// NeighborWiseAttention_2173253452653
// MI455X (gfx1250) — matrix operand order repaired and checked by reading, not compiled, not run on hardware
//
#include <hip/hip_runtime.h>

typedef _Float16 half_t;
typedef __attribute__((ext_vector_type(16))) _Float16 v16h;
typedef __attribute__((ext_vector_type(8)))  _Float16 v8h;
typedef __attribute__((ext_vector_type(8)))  float    v8f;

#define NNEI    120
#define ED      128
#define HD      128
#define NP      128              // padded neighbor count
#define HSTR    136              // f16 row stride in halves (272B, bank-conflict free)
#define SCALING 0.08838834764831845f   // 128^-0.5
#define LN_EPS  1e-5f
#define BIGNEG  (-1.0e30f)

// ---- LDS layout (bytes), all 16B aligned ----
#define OFF_XH   0               // x, f16 [128][HSTR]
#define OFF_WT   34816           // in_w^T f16 [384][HSTR]; reused for out_w^T
#define OFF_Q    139264          // q -> s -> o, f16 [128][HSTR]
#define OFF_K    174080          // k, f16 [128][HSTR]
#define OFF_VT   208896          // v^T (written transposed by GEMM1), f16 [128][HSTR]
#define OFF_BIAS 243712          // in_b, f32 [384]
#define OFF_RX   245248          // rx/ry/rz, f32 [3][128]
#define OFF_MASK 246784          // mask as f32 [128]
#define OFF_OB   247296          // out_b f32 [128]
#define OFF_LNG  247808          // ln gamma f32 [128]
#define OFF_LNB  248320          // ln beta f32 [128]
#define LDS_BYTES 248832

// A-fragment (16x32 f16): lane L<16 -> row rowbase+L, K = kbase+{0..7,16..23}
//                         lane L>=16 -> row rowbase+L-16, K = kbase+{8..15,24..31}
__device__ __forceinline__ v16h ldA(const half_t* base, int rowbase, int kbase, int lane) {
    int r  = rowbase + (lane & 15);
    int kb = kbase + ((lane & 16) ? 8 : 0);
    const half_t* p = base + r * HSTR + kb;
    v8h lo = *(const v8h*)(p);
    v8h hi = *(const v8h*)(p + 16);
    v16h a;
#pragma unroll
    for (int i = 0; i < 8; ++i) { a[i] = lo[i]; a[8 + i] = hi[i]; }
    return a;
}

// B-fragment (32x16 f16) from a transposed [n][k] LDS image:
// lane L<16 -> column n = nbase+L, K = kbase+{0..7,16..23} ; lane L>=16 -> K = kbase+{8..15,24..31}
__device__ __forceinline__ v16h ldB(const half_t* baseT, int nbase, int kbase, int lane) {
    int n  = nbase + (lane & 15);
    int kb = kbase + ((lane & 16) ? 8 : 0);
    const half_t* p = baseT + n * HSTR + kb;
    v8h lo = *(const v8h*)(p);
    v8h hi = *(const v8h*)(p + 16);
    v16h b;
#pragma unroll
    for (int i = 0; i < 8; ++i) { b[i] = lo[i]; b[8 + i] = hi[i]; }
    return b;
}

// Paired-output 16x16x128 step: two independent accumulator chains with
// one-K-chunk load lookahead so s_wait_dscnt never drains to 0 mid-stream.
__device__ __forceinline__ void gemm_pair(const v16h a[4], const half_t* baseT,
                                          int n0, int n1, int lane,
                                          v8f& acc0, v8f& acc1) {
    v16h b0 = ldB(baseT, n0, 0, lane);
    v16h c0 = ldB(baseT, n1, 0, lane);
#pragma unroll
    for (int kc = 0; kc < 4; ++kc) {
        v16h b1 = b0, c1 = c0;
        if (kc < 3) {
            b1 = ldB(baseT, n0, (kc + 1) * 32, lane);
            c1 = ldB(baseT, n1, (kc + 1) * 32, lane);
        }
        acc0 = __builtin_amdgcn_wmma_f32_16x16x32_f16(false, a[kc], false, b0,
                                                      (short)0, acc0, false, false);
        acc1 = __builtin_amdgcn_wmma_f32_16x16x32_f16(false, a[kc], false, c0,
                                                      (short)0, acc1, false, false);
        b0 = b1; c0 = c1;
    }
}

__global__ __launch_bounds__(256, 1)
void attn2_wmma_kernel(const float* __restrict__ G, const float* __restrict__ R,
                       const float* __restrict__ inw, const float* __restrict__ inb,
                       const float* __restrict__ outw, const float* __restrict__ outb,
                       const float* __restrict__ lng_g, const float* __restrict__ lnb_g,
                       const unsigned char* __restrict__ nmask, float* __restrict__ out)
{
    extern __shared__ __align__(16) char smem[];
    half_t* xh = (half_t*)(smem + OFF_XH);
    half_t* wt = (half_t*)(smem + OFF_WT);
    half_t* qs = (half_t*)(smem + OFF_Q);
    half_t* ks = (half_t*)(smem + OFF_K);
    half_t* vt = (half_t*)(smem + OFF_VT);
    float*  biasL = (float*)(smem + OFF_BIAS);
    float*  rx = (float*)(smem + OFF_RX);
    float*  ry = rx + 128;
    float*  rz = rx + 256;
    float*  mk = (float*)(smem + OFF_MASK);
    float*  ob = (float*)(smem + OFF_OB);
    float*  lng = (float*)(smem + OFF_LNG);
    float*  lnb = (float*)(smem + OFF_LNB);

    const int tid   = threadIdx.x;
    const int b     = blockIdx.x;
    const int lane  = tid & 31;
    const int wave  = tid >> 5;
    const int half_ = (lane >> 4);
    const int ln15  = lane & 15;
    const int rowbase = wave * 16;

    // ---------- one-time staging: x (f32->f16, pad rows zero), r, mask ----------
    const float* Gb = G + (size_t)b * (NNEI * ED);
    for (int idx = tid; idx < NP * ED; idx += 256) {
        int r = idx >> 7, c = idx & 127;
        float v = (r < NNEI) ? Gb[r * ED + c] : 0.f;
        xh[r * HSTR + c] = (half_t)v;
    }
    const float* Rb = R + (size_t)b * (NNEI * 3);
    if (tid < 128) {
        float x = 0.f, y = 0.f, z = 0.f;
        if (tid < NNEI) { x = Rb[tid * 3]; y = Rb[tid * 3 + 1]; z = Rb[tid * 3 + 2]; }
        rx[tid] = x; ry[tid] = y; rz[tid] = z;
        mk[tid] = (tid < NNEI && nmask[(size_t)b * NNEI + tid]) ? 1.f : 0.f;
    }

    for (int l = 0; l < 2; ++l) {
        // ---------- P0: stage in_w^T, biases, LN params ----------
        const float* W1 = inw + (size_t)l * ED * (3 * HD);
        for (int idx = tid; idx < ED * 3 * HD; idx += 256) {
            int kk = idx / 384, nn = idx - kk * 384;
            wt[nn * HSTR + kk] = (half_t)W1[idx];
        }
        for (int idx = tid; idx < 384; idx += 256) biasL[idx] = inb[l * 384 + idx];
        if (tid < 128) {
            ob[tid]  = outb[l * 128 + tid];
            lng[tid] = lng_g[l * 128 + tid];
            lnb[tid] = lnb_g[l * 128 + tid];
        }
        __syncthreads();

        // prefetch next layer's weights into cache while layer 0 computes
        if (l == 0) {
            const float* nw1 = inw + (size_t)ED * (3 * HD);
            for (int off = tid * 64; off < ED * 3 * HD; off += 256 * 64)
                __builtin_prefetch(nw1 + off, 0, 1);
            const float* nw2 = outw + (size_t)HD * ED;
            for (int off = tid * 64; off < HD * ED; off += 256 * 64)
                __builtin_prefetch(nw2 + off, 0, 1);
        }

        // ---------- P1: QKV = X * Wi + b  (wave owns 16 rows) ----------
        {
            v16h aX[4];
#pragma unroll
            for (int kc = 0; kc < 4; ++kc) aX[kc] = ldA(xh, rowbase, kc * 32, lane);
#pragma unroll
            for (int f = 0; f < 24; f += 2) {
                v8f acc0 = {}, acc1 = {};
                gemm_pair(aX, wt, f * 16, (f + 1) * 16, lane, acc0, acc1);
#pragma unroll
                for (int p = 0; p < 2; ++p) {
                    int fp = f + p;
                    v8f acc = p ? acc1 : acc0;
                    float bv = biasL[fp * 16 + ln15];
                    if (fp < 16) {
                        half_t* dst = (fp < 8) ? qs : ks;
                        int cb = (fp & 7) * 16;
#pragma unroll
                        for (int v = 0; v < 8; ++v) {
                            int rr = rowbase + v + 8 * half_;
                            dst[rr * HSTR + cb + ln15] = (half_t)(acc[v] + bv);
                        }
                    } else {
                        // v written TRANSPOSED: vt[d][i]; 8 consecutive rows -> one b128 store
                        int d = (fp - 16) * 16 + ln15;
                        v8h pack;
#pragma unroll
                        for (int v = 0; v < 8; ++v) pack[v] = (half_t)(acc[v] + bv);
                        *(v8h*)(vt + d * HSTR + rowbase + 8 * half_) = pack;
                    }
                }
            }
        }
        __syncthreads();

        // ---------- P2: row L2-normalize q/k (vectorized); v^T column-normalize; stage out_w^T ----------
        if (tid < 128) {
            int r = tid;
            v8h buf[16];
            float ss = 0.f;
#pragma unroll
            for (int c = 0; c < 16; ++c) {
                buf[c] = *(const v8h*)(qs + r * HSTR + c * 8);
#pragma unroll
                for (int i = 0; i < 8; ++i) { float f = (float)buf[c][i]; ss += f * f; }
            }
            float sc = SCALING / fmaxf(sqrtf(ss), 1e-12f);
#pragma unroll
            for (int c = 0; c < 16; ++c) {
                v8h o;
#pragma unroll
                for (int i = 0; i < 8; ++i) o[i] = (half_t)((float)buf[c][i] * sc);
                *(v8h*)(qs + r * HSTR + c * 8) = o;
            }
            // v column j = r of vt: strided walk
            float sv = 0.f;
            for (int d = 0; d < HD; ++d) { float f = (float)vt[d * HSTR + r]; sv += f * f; }
            float scv = 1.f / fmaxf(sqrtf(sv), 1e-12f);
            for (int d = 0; d < HD; ++d)
                vt[d * HSTR + r] = (half_t)((float)vt[d * HSTR + r] * scv);
        } else {
            int r = tid - 128;
            v8h buf[16];
            float ss = 0.f;
#pragma unroll
            for (int c = 0; c < 16; ++c) {
                buf[c] = *(const v8h*)(ks + r * HSTR + c * 8);
#pragma unroll
                for (int i = 0; i < 8; ++i) { float f = (float)buf[c][i]; ss += f * f; }
            }
            float sc = 1.f / fmaxf(sqrtf(ss), 1e-12f);
#pragma unroll
            for (int c = 0; c < 16; ++c) {
                v8h o;
#pragma unroll
                for (int i = 0; i < 8; ++i) o[i] = (half_t)((float)buf[c][i] * sc);
                *(v8h*)(ks + r * HSTR + c * 8) = o;
            }
        }
        // out_w^T into wt (GEMM1 done reading it)
        {
            const float* W2 = outw + (size_t)l * HD * ED;
            for (int idx = tid; idx < HD * ED; idx += 256) {
                int d = idx >> 7, e = idx & 127;
                wt[e * HSTR + d] = (half_t)W2[idx];
            }
        }
        __syncthreads();

        // ---------- P3: S = q k^T, mask, softmax, *ang -> s (f16, over own q rows) ----------
        {
            v16h aQ[4];
#pragma unroll
            for (int kc = 0; kc < 4; ++kc) aQ[kc] = ldA(qs, rowbase, kc * 32, lane);
            v8f S[8];
#pragma unroll
            for (int f = 0; f < 8; f += 2) {
                v8f a0 = {}, a1 = {};
                gemm_pair(aQ, ks, f * 16, (f + 1) * 16, lane, a0, a1);
                S[f] = a0; S[f + 1] = a1;
            }
            // column mask
#pragma unroll
            for (int f = 0; f < 8; ++f) {
                float mj = mk[f * 16 + ln15];
#pragma unroll
                for (int v = 0; v < 8; ++v) S[f][v] = (mj > 0.f) ? S[f][v] : BIGNEG;
            }
            // row max / exp / row sum (half-wave shuffle reductions)
            float rinv[8];
#pragma unroll
            for (int v = 0; v < 8; ++v) {
                float m = S[0][v];
#pragma unroll
                for (int f = 1; f < 8; ++f) m = fmaxf(m, S[f][v]);
#pragma unroll
                for (int x = 1; x < 16; x <<= 1) m = fmaxf(m, __shfl_xor(m, x, 16));
                float s = 0.f;
#pragma unroll
                for (int f = 0; f < 8; ++f) { float e = __expf(S[f][v] - m); S[f][v] = e; s += e; }
#pragma unroll
                for (int x = 1; x < 16; x <<= 1) s += __shfl_xor(s, x, 16);
                rinv[v] = 1.f / s;
            }
            // row mask + angular weights, store s into own q rows
            float rix[8], riy[8], riz[8], rmi[8];
#pragma unroll
            for (int v = 0; v < 8; ++v) {
                int i = rowbase + v + 8 * half_;
                rix[v] = rx[i]; riy[v] = ry[i]; riz[v] = rz[i]; rmi[v] = mk[i];
            }
#pragma unroll
            for (int f = 0; f < 8; ++f) {
                int j = f * 16 + ln15;
                float jx = rx[j], jy = ry[j], jz = rz[j];
#pragma unroll
                for (int v = 0; v < 8; ++v) {
                    float ang = rix[v] * jx + riy[v] * jy + riz[v] * jz;
                    float w = S[f][v] * rinv[v] * rmi[v] * ang;
                    int i = rowbase + v + 8 * half_;
                    qs[i * HSTR + j] = (half_t)w;
                }
            }
        }
        // no barrier needed: each wave reads only its own s rows next

        // ---------- P4: O = S * V  -> o (f16, over own rows) ----------
        {
            v16h aS[4];
#pragma unroll
            for (int kc = 0; kc < 4; ++kc) aS[kc] = ldA(qs, rowbase, kc * 32, lane);
            v8f O[8];
#pragma unroll
            for (int dt = 0; dt < 8; dt += 2) {
                v8f a0 = {}, a1 = {};
                gemm_pair(aS, vt, dt * 16, (dt + 1) * 16, lane, a0, a1);
                O[dt] = a0; O[dt + 1] = a1;
            }
#pragma unroll
            for (int dt = 0; dt < 8; ++dt)
#pragma unroll
                for (int v = 0; v < 8; ++v) {
                    int i = rowbase + v + 8 * half_;
                    qs[i * HSTR + dt * 16 + ln15] = (half_t)O[dt][v];
                }
        }

        // ---------- P5: OUT = O * Wo + bo + residual; layernorm ----------
        {
            v16h aO[4];
#pragma unroll
            for (int kc = 0; kc < 4; ++kc) aO[kc] = ldA(qs, rowbase, kc * 32, lane);
            v8f F[8];
#pragma unroll
            for (int et = 0; et < 8; et += 2) {
                v8f a0 = {}, a1 = {};
                gemm_pair(aO, wt, et * 16, (et + 1) * 16, lane, a0, a1);
                F[et] = a0; F[et + 1] = a1;
            }
#pragma unroll
            for (int et = 0; et < 8; ++et) {
                int e = et * 16 + ln15;
                float bo = ob[e];
#pragma unroll
                for (int v = 0; v < 8; ++v) {
                    int i = rowbase + v + 8 * half_;
                    F[et][v] += bo + (float)xh[i * HSTR + e];
                }
            }
            float mu[8], rv[8];
#pragma unroll
            for (int v = 0; v < 8; ++v) {
                float s = 0.f;
#pragma unroll
                for (int et = 0; et < 8; ++et) s += F[et][v];
#pragma unroll
                for (int x = 1; x < 16; x <<= 1) s += __shfl_xor(s, x, 16);
                mu[v] = s * (1.f / 128.f);
                float q = 0.f;
#pragma unroll
                for (int et = 0; et < 8; ++et) { float d = F[et][v] - mu[v]; q += d * d; }
#pragma unroll
                for (int x = 1; x < 16; x <<= 1) q += __shfl_xor(q, x, 16);
                rv[v] = rsqrtf(q * (1.f / 128.f) + LN_EPS);
            }
#pragma unroll
            for (int et = 0; et < 8; ++et) {
                int e = et * 16 + ln15;
                float g = lng[e], bb2 = lnb[e];
#pragma unroll
                for (int v = 0; v < 8; ++v) {
                    int i = rowbase + v + 8 * half_;
                    float o = (F[et][v] - mu[v]) * rv[v] * g + bb2;
                    if (l == 1) {
                        if (i < NNEI) out[(size_t)b * (NNEI * ED) + i * ED + e] = o;
                    } else {
                        xh[i * HSTR + e] = (half_t)o;
                    }
                }
            }
        }
        __syncthreads();   // before next layer's weight re-staging
    }
}

extern "C" void kernel_launch(void* const* d_in, const int* in_sizes, int n_in,
                              void* d_out, int out_size, void* d_ws, size_t ws_size,
                              hipStream_t stream) {
    const float* G    = (const float*)d_in[0];
    const float* R    = (const float*)d_in[1];
    const float* inw  = (const float*)d_in[2];
    const float* inb  = (const float*)d_in[3];
    const float* outw = (const float*)d_in[4];
    const float* outb = (const float*)d_in[5];
    const float* lng  = (const float*)d_in[6];
    const float* lnb  = (const float*)d_in[7];
    const unsigned char* nmask = (const unsigned char*)d_in[8];
    float* out = (float*)d_out;

    hipFuncSetAttribute((const void*)attn2_wmma_kernel,
                        hipFuncAttributeMaxDynamicSharedMemorySize, LDS_BYTES);
    attn2_wmma_kernel<<<8192, 256, LDS_BYTES, stream>>>(G, R, inw, inb, outw, outb,
                                                        lng, lnb, nmask, out);
}
